// Conv_SQT_22213570855264
// MI455X (gfx1250) — matrix operand order repaired and checked by reading, not compiled, not run on hardware
//
#include <hip/hip_runtime.h>

typedef __bf16 bf16;
typedef __attribute__((ext_vector_type(16))) __bf16 v16bf;
typedef __attribute__((ext_vector_type(8)))  float  v8f;

union FragU {
    uint4 u[2];
    v16bf v;
};

// ---------------------------------------------------------------------------
// Implicit-GEMM convolution using WMMA bf16 -> f32, f32 accumulate.
//   MODE 0: normal conv, stride 1, pad KSZ/2  (HIN==HOUT, WIN==WOUT)
//   MODE 1: ConvTranspose2d stride2/pad1/outpad1 as conv over 2x-dilated input
//   EPI  0: ReLU        EPI 1: tanh
// Block = WAVES*32 threads; MTILE = WAVES*16 == COUT (whole channel dim per
// workgroup, so the im2col B-tile is gathered exactly once).
// Grid: x = (HOUT*WOUT)/64 N-tiles, y = 1, z = batch.
// All spatial dims are compile-time powers of two -> shift/mask addressing.
// LDS ping-pong: one barrier per K-step; next-step gather issues before the
// current step's WMMAs so global latency hides under matrix math.
// ---------------------------------------------------------------------------
template <int CIN, int COUT, int KSZ, int MODE, int EPI, int WAVES,
          int HOUT, int WOUT, int HIN, int WIN>
__global__ __launch_bounds__(WAVES * 32)
void conv_wmma_kernel(const bf16* __restrict__ in, const bf16* __restrict__ w,
                      const float* __restrict__ bias, bf16* __restrict__ out) {
    static_assert(WAVES * 16 == COUT, "MTILE must equal COUT");
    constexpr int KDIM  = CIN * KSZ * KSZ;     // GEMM K (multiple of 32)
    constexpr int NSTEP = KDIM / 32;
    constexpr int PAD   = KSZ / 2;
    constexpr int LDSW  = 40;                  // padded row stride (bf16)

    __shared__ alignas(16) bf16 ldsB[2][64][LDSW];

    const int tid   = threadIdx.x;
    const int lane  = tid & 31;
    const int wave  = tid >> 5;
    const int b     = blockIdx.z;
    const int n0    = blockIdx.x * 64;         // N tile base (pixel index)
    const int mwave = wave * 16;

    // gather role: fixed k within the 32-wide K-step, sweep pixels
    const int kf    = tid & 31;
    const int nbase = tid >> 5;

    const size_t in_b = (size_t)b * CIN * HIN * WIN;

    auto fill = [&](int buf, int ks) {
        const int kk = ks * 32 + kf;
        const int ci = kk / (KSZ * KSZ);
        const int rr = kk - ci * (KSZ * KSZ);
        const int kh = rr / KSZ;
        const int kw = rr - kh * KSZ;
        const bf16* ip = in + in_b + (size_t)ci * HIN * WIN;
        #pragma unroll
        for (int j = 0; j < 64 / WAVES; ++j) {
            const int n  = nbase + j * WAVES;
            const int np = n0 + n;
            const int y  = np / WOUT;          // WOUT is power of two
            const int x  = np & (WOUT - 1);
            bf16 val = (bf16)0.0f;
            if (MODE == 0) {
                const int iy = y + kh - PAD;
                const int ix = x + kw - PAD;
                if (iy >= 0 && iy < HIN && ix >= 0 && ix < WIN)
                    val = ip[iy * WIN + ix];
            } else {
                const int iy = y + kh - 1;
                const int ix = x + kw - 1;
                if (iy >= 0 && ix >= 0 && !(iy & 1) && !(ix & 1) &&
                    (iy >> 1) < HIN && (ix >> 1) < WIN)
                    val = ip[(iy >> 1) * WIN + (ix >> 1)];
            }
            ldsB[buf][n][kf] = val;
        }
    };

    v8f acc[4] = {};

    fill(0, 0);
    __syncthreads();

    for (int ks = 0; ks < NSTEP; ++ks) {
        const int cur = ks & 1;
        // prefetch next B tile into the other LDS buffer (loads issue first)
        if (ks + 1 < NSTEP) fill(cur ^ 1, ks + 1);

        // A fragment straight from global (ISA 16-bit A 16x32 layout)
        FragU fa;
        {
            const int m   = mwave + (lane & 15);
            const int loK = (lane & 16) ? 8 : 0;  // lanes 16-31: K 8-15 / 24-31
            const uint4* p = (const uint4*)(w + (size_t)m * KDIM + ks * 32 + loK);
            fa.u[0] = p[0];    // K lo .. lo+7
            fa.u[1] = p[2];    // K lo+16 .. lo+23
        }

        #pragma unroll
        for (int t = 0; t < 4; ++t) {
            FragU fb;
            const int col = t * 16 + (lane & 15);
            const uint4* q = (const uint4*)(&ldsB[cur][col][(lane >> 4) * 8]);
            fb.u[0] = q[0];
            fb.u[1] = q[2];
            acc[t] = __builtin_amdgcn_wmma_f32_16x16x32_bf16(
                false, fa.v, false, fb.v, (short)0, acc[t], false, false);
        }
        __syncthreads();
    }

    // epilogue: bias + activation, store bf16 (C/D VGPR layout)
    const int moff = (lane & 16) ? 8 : 0;
    const int col  = lane & 15;
    #pragma unroll
    for (int t = 0; t < 4; ++t) {
        const int n = n0 + t * 16 + col;
        const int y = n / WOUT;
        const int x = n & (WOUT - 1);
        #pragma unroll
        for (int r = 0; r < 8; ++r) {
            const int m = mwave + moff + r;
            float v = acc[t][r];
            if (bias) v += bias[m];
            v = (EPI == 0) ? fmaxf(v, 0.0f) : tanhf(v);
            out[((size_t)(b * COUT + m) * HOUT + y) * WOUT + x] = (bf16)v;
        }
    }
}

// ---------------------------------------------------------------------------
// Routed conv1 (6->128, 3x3, pad 1) + ReLU + 2x2 mean pool, fused. f32 math,
// bf16 output. One thread per pooled output element. ~3.6 GFLOP (<1% total).
// ---------------------------------------------------------------------------
__global__ __launch_bounds__(256)
void conv1_pool_kernel(const float* __restrict__ in0, const float* __restrict__ in1,
                       const int* __restrict__ route, const float* __restrict__ w,
                       const float* __restrict__ bias, bf16* __restrict__ out) {
    const int idx = blockIdx.x * blockDim.x + threadIdx.x;
    if (idx >= 64 * 128 * 32 * 32) return;
    const int x = idx & 31;
    const int y = (idx >> 5) & 31;
    const int c = (idx >> 10) & 127;
    const int b = idx >> 17;
    const int e = route[b];
    const float* wb = w + ((size_t)e * 128 + c) * 54;
    const float  bv = bias[e * 128 + c];
    float s = 0.0f;
    for (int dy = 0; dy < 2; ++dy)
    for (int dx = 0; dx < 2; ++dx) {
        const int oy = 2 * y + dy, ox = 2 * x + dx;
        float a = bv;
        for (int ci = 0; ci < 6; ++ci) {
            const float* ip = (ci < 3) ? in0 + ((size_t)b * 3 + ci) * 4096
                                       : in1 + ((size_t)b * 3 + (ci - 3)) * 4096;
            const float* wp = wb + ci * 9;
            #pragma unroll
            for (int kh = 0; kh < 3; ++kh) {
                const int iy = oy + kh - 1;
                if (iy < 0 || iy >= 64) continue;
                #pragma unroll
                for (int kw = 0; kw < 3; ++kw) {
                    const int ix = ox + kw - 1;
                    if (ix < 0 || ix >= 64) continue;
                    a += ip[iy * 64 + ix] * wp[kh * 3 + kw];
                }
            }
        }
        s += fmaxf(a, 0.0f);
    }
    out[idx] = (bf16)(s * 0.25f);
}

// ---------------------------------------------------------------------------
// Final routed transpose-conv (stride 1 == conv with flipped/transposed
// kernel), 128->9, 3x3, pad 1. Writes tuple-concatenated f32 output.
// ---------------------------------------------------------------------------
__global__ __launch_bounds__(256)
void convup1_routed_kernel(const bf16* __restrict__ act, const int* __restrict__ route,
                           const float* __restrict__ w, const float* __restrict__ bias,
                           float* __restrict__ out) {
    const int idx = blockIdx.x * blockDim.x + threadIdx.x;
    if (idx >= 64 * 9 * 4096) return;
    const int x = idx & 63;
    const int y = (idx >> 6) & 63;
    const int c = (idx >> 12) % 9;
    const int b = idx / (9 * 4096);
    const int e = route[b];
    float a = bias[e * 9 + c];
    for (int ci = 0; ci < 128; ++ci) {
        const bf16*  ap = act + ((size_t)b * 128 + ci) * 4096;
        const float* wp = w + (((size_t)e * 128 + ci) * 9 + c) * 9;
        #pragma unroll
        for (int kh = 0; kh < 3; ++kh) {
            const int iy = y + kh - 1;
            if (iy < 0 || iy >= 64) continue;
            #pragma unroll
            for (int kw = 0; kw < 3; ++kw) {
                const int ix = x + kw - 1;
                if (ix < 0 || ix >= 64) continue;
                a += (float)ap[iy * 64 + ix] * wp[(2 - kh) * 3 + (2 - kw)];
            }
        }
    }
    const int chunk = c / 3, cc = c - chunk * 3;
    out[(size_t)chunk * (64 * 3 * 4096) + ((size_t)b * 3 + cc) * 4096 + y * 64 + x] = a;
}

// ---------------------------------------------------------------------------
// Weight prep: f32 -> bf16 (plain), and flip+transpose for ConvTranspose2d.
// ---------------------------------------------------------------------------
__global__ void cvt_bf16_kernel(const float* __restrict__ src, bf16* __restrict__ dst, int n) {
    const int i = blockIdx.x * blockDim.x + threadIdx.x;
    if (i < n) dst[i] = (bf16)src[i];
}

__global__ void cvt_up2_kernel(const float* __restrict__ src, bf16* __restrict__ dst) {
    // src: [256][128][3][3] (cin, cout, kh, kw) -> dst: [128][256][3][3] flipped
    const int i = blockIdx.x * blockDim.x + threadIdx.x;
    if (i >= 128 * 256 * 9) return;
    const int kw = i % 3;
    const int kh = (i / 3) % 3;
    const int ci = (i / 9) % 256;
    const int co = i / (9 * 256);
    dst[i] = (bf16)src[((size_t)(ci * 128 + co) * 3 + (2 - kh)) * 3 + (2 - kw)];
}

// ---------------------------------------------------------------------------
extern "C" void kernel_launch(void* const* d_in, const int* in_sizes, int n_in,
                              void* d_out, int out_size, void* d_ws, size_t ws_size,
                              hipStream_t stream) {
    const float* in0 = (const float*)d_in[0];
    const float* in1 = (const float*)d_in[1];
    const int*   ri  = (const int*)d_in[2];
    const int*   ro  = (const int*)d_in[3];
    const float* c1w = (const float*)d_in[4];
    const float* c1b = (const float*)d_in[5];
    const float* c2w = (const float*)d_in[6];
    const float* m1w = (const float*)d_in[7];
    const float* m1b = (const float*)d_in[8];
    const float* m2w = (const float*)d_in[9];
    const float* m2b = (const float*)d_in[10];
    const float* u2w = (const float*)d_in[11];
    const float* u2b = (const float*)d_in[12];
    const float* u1w = (const float*)d_in[13];
    const float* u1b = (const float*)d_in[14];
    float* out = (float*)d_out;

    char* ws = (char*)d_ws;
    size_t off = 0;
    auto alloc = [&](size_t bytes) -> void* {
        void* p = ws + off;
        off += (bytes + 255) & ~(size_t)255;
        return p;
    };
    bf16* actA = (bf16*)alloc((size_t)64 * 128 * 32 * 32 * 2);  // conv1+pool out
    bf16* actB = (bf16*)alloc((size_t)64 * 256 * 32 * 32 * 2);  // conv2 / mid2 out
    bf16* actC = (bf16*)alloc((size_t)64 * 256 * 32 * 32 * 2);  // mid1 out
    bf16* actD = (bf16*)alloc((size_t)64 * 128 * 64 * 64 * 2);  // convup2 out
    bf16* w2   = (bf16*)alloc((size_t)256 * 128 * 9 * 2);
    bf16* wm1  = (bf16*)alloc((size_t)256 * 256 * 25 * 2);
    bf16* wm2  = (bf16*)alloc((size_t)256 * 256 * 25 * 2);
    bf16* wu2  = (bf16*)alloc((size_t)128 * 256 * 9 * 2);

    // weight prep
    {
        int n = 256 * 128 * 9;
        cvt_bf16_kernel<<<(n + 255) / 256, 256, 0, stream>>>(c2w, w2, n);
        n = 256 * 256 * 25;
        cvt_bf16_kernel<<<(n + 255) / 256, 256, 0, stream>>>(m1w, wm1, n);
        cvt_bf16_kernel<<<(n + 255) / 256, 256, 0, stream>>>(m2w, wm2, n);
        n = 128 * 256 * 9;
        cvt_up2_kernel<<<(n + 255) / 256, 256, 0, stream>>>(u2w, wu2);
    }

    // routed conv1 + relu + 2x2 mean pool (scalar f32)
    {
        int n = 64 * 128 * 32 * 32;
        conv1_pool_kernel<<<(n + 255) / 256, 256, 0, stream>>>(in0, in1, ri, c1w, c1b, actA);
    }

    // conv2: 128->256, 3x3, relu, no bias       (WMMA, 16 waves, MTILE=256)
    {
        dim3 g(1024 / 64, 1, 64);
        conv_wmma_kernel<128, 256, 3, 0, 0, 16, 32, 32, 32, 32>
            <<<g, 512, 0, stream>>>(actA, w2, nullptr, actB);
    }
    // mid1: 256->256, 5x5, +bias, relu          (WMMA)
    {
        dim3 g(1024 / 64, 1, 64);
        conv_wmma_kernel<256, 256, 5, 0, 0, 16, 32, 32, 32, 32>
            <<<g, 512, 0, stream>>>(actB, wm1, m1b, actC);
    }
    // mid2: 256->256, 5x5, +bias, relu          (WMMA)
    {
        dim3 g(1024 / 64, 1, 64);
        conv_wmma_kernel<256, 256, 5, 0, 0, 16, 32, 32, 32, 32>
            <<<g, 512, 0, stream>>>(actC, wm2, m2b, actB);
    }
    // convup2: transpose conv stride2 as dilated implicit GEMM, +bias, tanh
    {
        dim3 g(4096 / 64, 1, 64);
        conv_wmma_kernel<256, 128, 3, 1, 1, 8, 64, 64, 32, 32>
            <<<g, 256, 0, stream>>>(actB, wu2, u2b, actD);
    }
    // routed convup1 -> tuple-concatenated f32 output
    {
        int n = 64 * 9 * 4096;
        convup1_routed_kernel<<<(n + 255) / 256, 256, 0, stream>>>(actD, ro, u1w, u1b, out);
    }
}
